// BahdanauAttention_42374147342422
// MI455X (gfx1250) — matrix operand order repaired and checked by reading, not compiled, not run on hardware
//
#include <hip/hip_runtime.h>

// ---------------------------------------------------------------------------
// Bahdanau attention w/ coverage, fused for MI455X (gfx1250, wave32, WMMA).
// B=32, S=2048, D_ENC=D_DEC=UNITS=1024.
// ---------------------------------------------------------------------------

typedef __attribute__((ext_vector_type(16))) __bf16        v16bf;
typedef __attribute__((ext_vector_type(8)))  float         v8f;
typedef __attribute__((ext_vector_type(4)))  float         v4f;
typedef __attribute__((ext_vector_type(4)))  unsigned int  v4u;

#define NB     32      // batch
#define NS     2048    // seq len
#define ND     1024    // d_enc = d_dec = units
#define SROWS  64      // s-rows (GEMM M) per workgroup
#define LDA    1032    // LDS A-tile row stride in bf16 elems (even, padded)

__device__ __forceinline__ unsigned short f2bf(float x) {
  unsigned int u = __float_as_uint(x);
  u += 0x7FFFu + ((u >> 16) & 1u);   // round-to-nearest-even
  return (unsigned short)(u >> 16);
}

// Branch-free tanh: 1 - 2/(e^{2x}+1).  One v_exp_f32 + one v_rcp_f32.
// e->inf => 1-0 = 1 ; e->0 => 1-2 = -1  (correct saturation, no NaN).
__device__ __forceinline__ float fast_tanh(float x) {
  float e = __expf(2.0f * x);
  return 1.0f - 2.0f * __builtin_amdgcn_rcpf(e + 1.0f);
}

// ---------------------------------------------------------------------------
// Kernel 1: Ut[u][d] = bf16(U[d][u]).  LDS tiled transpose. grid (32,32)x256.
// ---------------------------------------------------------------------------
__global__ __launch_bounds__(256) void k_prep_ut(const float* __restrict__ U,
                                                 unsigned short* __restrict__ Ut) {
  __shared__ unsigned short tile[32][33];
  int bx = blockIdx.x;            // d tile
  int by = blockIdx.y;            // u tile
  int tx = threadIdx.x & 31;
  int ty = threadIdx.x >> 5;      // 0..7
  for (int i = ty; i < 32; i += 8) {
    int d = bx * 32 + i;
    int u = by * 32 + tx;
    tile[i][tx] = f2bf(U[(size_t)d * ND + u]);
  }
  __syncthreads();
  for (int i = ty; i < 32; i += 8) {
    int u = by * 32 + i;
    int d = bx * 32 + tx;
    Ut[(size_t)u * ND + d] = tile[tx][i];
  }
}

// ---------------------------------------------------------------------------
// Kernel 2: base[b][u] = sum_d h[b][d]*W[d][u] + Wc_b[u].  grid 32 x 256.
// ---------------------------------------------------------------------------
__global__ __launch_bounds__(256) void k_base(const float* __restrict__ h,
                                              const float* __restrict__ W,
                                              const float* __restrict__ Wcb,
                                              float* __restrict__ base) {
  int b = blockIdx.x;
  for (int u = threadIdx.x; u < ND; u += 256) {
    float acc = Wcb[u];
    for (int d = 0; d < ND; ++d)
      acc += h[b * ND + d] * W[(size_t)d * ND + u];
    base[b * ND + u] = acc;
  }
}

// ---------------------------------------------------------------------------
// Kernel 3 (main): fused  score[b,s] = sum_u tanh( (enc@U)[b,s,u] + base[b,u]
//                                               + cov[b,s]*Wc_w[u] ) * V[u]
// One WG = 64 s-rows of one batch. 8 waves, each wave owns 128 N-columns and
// 4 M-tiles of 16 rows; K accumulated in 32-wide bf16 WMMA steps.
// grid = 32 * (2048/64) = 1024 blocks x 256 threads.
// ---------------------------------------------------------------------------
__global__ __launch_bounds__(256) void k_scores(const float* __restrict__ enc,
                                                const float* __restrict__ cov,
                                                const float* __restrict__ base,
                                                const unsigned short* __restrict__ Ut,
                                                const float* __restrict__ Wcw,
                                                const float* __restrict__ Vv,
                                                float* __restrict__ scores) {
  __shared__ unsigned short Alds[SROWS * LDA];   // 64 x 1024 bf16 A tile
  __shared__ float covS[SROWS];
  __shared__ float scoreW[8][SROWS];             // per-wave partial scores

  int b  = blockIdx.x >> 5;                      // NS/SROWS = 32 tiles per batch
  int st = blockIdx.x & 31;
  int s0 = st * SROWS;
  int tid  = threadIdx.x;
  int lane = tid & 31;
  int wave = tid >> 5;
  int l15  = lane & 15;
  int kh   = lane >> 4;

  // ---- Phase 1: enc rows -> bf16 in LDS (coalesced float4 loads) ----
  const float* encb = enc + ((size_t)(b * NS + s0)) * ND;
  for (int c = tid; c < SROWS * (ND / 4); c += 256) {
    int row = c >> 8;                            // ND/4 = 256 chunks per row
    int c4  = c & 255;
    v4f v = *(const v4f*)(encb + (size_t)row * ND + c4 * 4);
    unsigned int p0 = (unsigned int)f2bf(v.x) | ((unsigned int)f2bf(v.y) << 16);
    unsigned int p1 = (unsigned int)f2bf(v.z) | ((unsigned int)f2bf(v.w) << 16);
    unsigned int* dst = (unsigned int*)&Alds[row * LDA + c4 * 4];
    dst[0] = p0;
    dst[1] = p1;
  }
  if (tid < SROWS) covS[tid] = cov[(size_t)b * NS + s0 + tid];
  __syncthreads();

  // ---- Phase 2: WMMA GEMM + fused tanh/V epilogue ----
  float part[4][8];
#pragma unroll
  for (int mt = 0; mt < 4; ++mt)
#pragma unroll
    for (int r = 0; r < 8; ++r) part[mt][r] = 0.0f;

  for (int nti = 0; nti < 8; ++nti) {
    int n0 = (wave * 8 + nti) * 16;
    int n  = n0 + l15;                              // this lane's N column
    const unsigned short* bp = Ut + (size_t)n * ND + kh * 8;

    v8f acc[4] = {{}, {}, {}, {}};

    for (int kb = 0; kb < 32; ++kb) {
      int k0 = kb * 32;

      // B fragment: Ut row n at K = k0 + kh*8 + {0..7, 16..23}
      union { v4u q[2]; v16bf v; } bu;
      bu.q[0] = *(const v4u*)(bp + k0);
      bu.q[1] = *(const v4u*)(bp + k0 + 16);

#pragma unroll
      for (int mt = 0; mt < 4; ++mt) {
        int row = mt * 16 + l15;
        // A fragment per ISA layout: lanes<16 K=k0+{0..7,16..23}, lanes>=16 +8
        const unsigned int* ap =
            (const unsigned int*)Alds + ((row * LDA + k0 + kh * 8) >> 1);
        union { unsigned int u[8]; v16bf v; } au;
        au.u[0] = ap[0];  au.u[1] = ap[1];  au.u[2] = ap[2];  au.u[3] = ap[3];
        au.u[4] = ap[8];  au.u[5] = ap[9];  au.u[6] = ap[10]; au.u[7] = ap[11];

        acc[mt] = __builtin_amdgcn_wmma_f32_16x16x32_bf16(
            false, au.v, false, bu.v, (short)0, acc[mt], false, false);
      }
    }

    // Epilogue: C element r of acc[mt] sits at M = mt*16 + r + 8*kh, N = n
    float bn = base[b * ND + n];
    float wc = Wcw[n];
    float vn = Vv[n];
#pragma unroll
    for (int mt = 0; mt < 4; ++mt) {
#pragma unroll
      for (int r = 0; r < 8; ++r) {
        int m = mt * 16 + r + 8 * kh;
        float t = fast_tanh(acc[mt][r] + bn + covS[m] * wc);
        part[mt][r] += t * vn;
      }
    }
  }

  // ---- Cross-lane (16 lanes share one M row) then cross-wave reduction ----
#pragma unroll
  for (int mt = 0; mt < 4; ++mt) {
#pragma unroll
    for (int r = 0; r < 8; ++r) {
      float v = part[mt][r];
      v += __shfl_xor(v, 1, 16);
      v += __shfl_xor(v, 2, 16);
      v += __shfl_xor(v, 4, 16);
      v += __shfl_xor(v, 8, 16);
      if (l15 == 0) scoreW[wave][mt * 16 + r + 8 * kh] = v;
    }
  }
  __syncthreads();
  if (tid < SROWS) {
    float s = 0.0f;
#pragma unroll
    for (int w = 0; w < 8; ++w) s += scoreW[w][tid];
    scores[(size_t)b * NS + s0 + tid] = s;
  }
}

// ---------------------------------------------------------------------------
// Kernel 4: masked softmax over S per batch + coverage_out. grid 32 x 256.
// softmax(score)*mask renormalized == exp(score-max)*mask / sum(...)
// ---------------------------------------------------------------------------
__global__ __launch_bounds__(256) void k_softmax(const float* __restrict__ scores,
                                                 const float* __restrict__ mask,
                                                 const float* __restrict__ cov,
                                                 float* __restrict__ attn,
                                                 float* __restrict__ covout) {
  __shared__ float red[256];
  int b = blockIdx.x, tid = threadIdx.x;
  const float* sc = scores + (size_t)b * NS;
  const float* mk = mask + (size_t)b * NS;

  float mx = -3.402823e38f;
  for (int s = tid; s < NS; s += 256) mx = fmaxf(mx, sc[s]);
  red[tid] = mx;
  __syncthreads();
  for (int off = 128; off > 0; off >>= 1) {
    if (tid < off) red[tid] = fmaxf(red[tid], red[tid + off]);
    __syncthreads();
  }
  mx = red[0];
  __syncthreads();

  float sum = 0.0f;
  for (int s = tid; s < NS; s += 256) {
    float e = __expf(sc[s] - mx) * mk[s];
    attn[(size_t)b * NS + s] = e;
    sum += e;
  }
  red[tid] = sum;
  __syncthreads();
  for (int off = 128; off > 0; off >>= 1) {
    if (tid < off) red[tid] += red[tid + off];
    __syncthreads();
  }
  float inv = 1.0f / red[0];

  for (int s = tid; s < NS; s += 256) {
    float w = attn[(size_t)b * NS + s] * inv;
    attn[(size_t)b * NS + s] = w;
    covout[(size_t)b * NS + s] = cov[(size_t)b * NS + s] + w;
  }
}

// ---------------------------------------------------------------------------
// Kernel 5: ctx[b][d] = sum_s w[b][s] * enc[b][s][d].
// grid 32*4 blocks x 256 threads; weights cached in LDS; coalesced enc reads.
// ---------------------------------------------------------------------------
__global__ __launch_bounds__(256) void k_context(const float* __restrict__ attn,
                                                 const float* __restrict__ enc,
                                                 float* __restrict__ ctx) {
  __shared__ float wS[NS];
  int b  = blockIdx.x >> 2;
  int ch = blockIdx.x & 3;
  int tid = threadIdx.x;
  for (int s = tid; s < NS; s += 256) wS[s] = attn[(size_t)b * NS + s];
  __syncthreads();
  int d = ch * 256 + tid;
  float acc = 0.0f;
  for (int s = 0; s < NS; ++s)
    acc += wS[s] * enc[((size_t)(b * NS + s)) * ND + d];
  ctx[b * ND + d] = acc;
}

// ---------------------------------------------------------------------------
extern "C" void kernel_launch(void* const* d_in, const int* in_sizes, int n_in,
                              void* d_out, int out_size, void* d_ws, size_t ws_size,
                              hipStream_t stream) {
  const float* enc  = (const float*)d_in[0];   // (32,2048,1024)
  const float* mask = (const float*)d_in[1];   // (32,2048)
  const float* h    = (const float*)d_in[2];   // (32,1024)
  const float* cov  = (const float*)d_in[3];   // (32,2048,1,1)
  const float* W    = (const float*)d_in[4];   // (1024,1024)
  const float* U    = (const float*)d_in[5];   // (1024,1024)
  const float* Wcw  = (const float*)d_in[6];   // (1,1024)
  const float* Wcb  = (const float*)d_in[7];   // (1024,)
  const float* Vv   = (const float*)d_in[8];   // (1024,1)

  float* out    = (float*)d_out;
  float* ctx    = out;                         // (32,1024)
  float* attn   = out + NB * ND;               // (32,2048)
  float* covout = attn + NB * NS;              // (32,2048,1,1)

  unsigned short* Ut = (unsigned short*)d_ws;                          // 2 MB
  float* base   = (float*)((char*)d_ws + (size_t)ND * ND * 2);         // 128 KB
  float* scores = base + NB * ND;                                      // 256 KB

  k_prep_ut<<<dim3(32, 32), 256, 0, stream>>>(U, Ut);
  k_base<<<NB, 256, 0, stream>>>(h, W, Wcb, base);
  k_scores<<<NB * (NS / SROWS), 256, 0, stream>>>(enc, cov, base, Ut, Wcw, Vv, scores);
  k_softmax<<<NB, 256, 0, stream>>>(scores, mask, cov, attn, covout);
  k_context<<<NB * 4, 256, 0, stream>>>(attn, enc, ctx);
}
